// Model_67422396613183
// MI455X (gfx1250) — matrix operand order repaired and checked by reading, not compiled, not run on hardware
//
#include <hip/hip_runtime.h>
#include <hip/hip_bf16.h>

// ---------------------------------------------------------------------------
// MI455X (gfx1250) implementation.
//
// Roofline: the 4-layer MLP streams ~1.07 GB of fp32 weights once (672-row
// activations stay resident in the 192 MB L2), so the net is HBM-bound:
// ~46 us at 23.3 TB/s vs ~363 GFLOP of math (needs only ~8 TF/s).  Weights
// are streamed with fully-coalesced float4 loads (all in flight before a
// single wait), packed to bf16 with v_cvt_pk_bf16_f32 into LDS, and fed to
// v_wmma_f32_16x16x32_bf16 with f32 accumulation.
// ---------------------------------------------------------------------------

#define B_  32
#define L_  512
#define C_  21
#define P_  256
#define R_  (B_ * C_)          // 672 rows for every batched GEMM
#define NPC (B_ * P_ * C_)     // 172032

typedef __attribute__((ext_vector_type(16))) __bf16 bf16x16;
typedef __attribute__((ext_vector_type(8)))  __bf16 bf16x8;
typedef __attribute__((ext_vector_type(4)))  __bf16 bf16x4;
typedef __attribute__((ext_vector_type(2)))  __bf16 bf16x2;
typedef __attribute__((ext_vector_type(8)))  float  f32x8;

enum { EPI_NONE = 0, EPI_GELU = 1, EPI_SOFTPLUS5 = 2, EPI_RESID_SQ = 3 };

__device__ __forceinline__ float gelu_tanh(float v) {
    // jax.nn.gelu (approximate=True)
    float u = 0.7978845608028654f * (v + 0.044715f * v * v * v);
    return 0.5f * v * (1.0f + tanhf(u));
}
__device__ __forceinline__ float softplus_f(float x) {
    // numerically stable softplus
    return fmaxf(x, 0.0f) + log1pf(expf(-fabsf(x)));
}

// ---------------------------------------------------------------------------
// WMMA GEMM:  Out[R x N] = epilogue( A[R x K] @ W[K x N] + bias )
// block = 256 threads (8 wave32).  WG tile: M=32, N=256.
// Each wave computes a 16(M) x 64(N) strip as 4 WMMA accumulators.
// K loop in steps of 32 (manually 2x unrolled -> compile-time LDS buffer
// indices), double-buffered LDS, bf16 staging.
// Requires: K % 64 == 0, N % 256 == 0, rows % 32 == 0 (true for all calls).
// ---------------------------------------------------------------------------
template <int EPI>
__global__ __launch_bounds__(256) void
gemm_bf16_wmma(const float* __restrict__ A, const float* __restrict__ W,
               const float* __restrict__ bias, float* __restrict__ Out,
               int K, int N,
               const float* __restrict__ resid,   // EPI_RESID_SQ only
               float* __restrict__ partial)       // EPI_RESID_SQ only
{
    // A tile: 32 rows x 32 k, k-contiguous (matches A-fragment lane layout).
    // B tile: stored transposed: Bs[col][k] so a B fragment (col = lane%16,
    // K = 8*hl + e, e<8; 8*hl + 8 + e, e>=8) is two 8-bf16 runs = two 16B ds loads.
    __shared__ __bf16 As[2][32][32];
    __shared__ __bf16 Bs[2][256][32];
    __shared__ float  red[256];

    const int t    = threadIdx.x;
    const int wave = t >> 5;
    const int lane = t & 31;
    const int wm   = wave >> 2;      // 0..1  (M sub-tile)
    const int wn   = wave & 3;       // 0..3  (N sub-tile of 64)
    const int hl   = lane >> 4;      // half-wave: 0 or 1
    const int r16  = lane & 15;
    const int m0   = blockIdx.y * 32;
    const int n0   = blockIdx.x * 256;

    f32x8 acc[4];
    #pragma unroll
    for (int j = 0; j < 4; ++j)
        #pragma unroll
        for (int e = 0; e < 8; ++e) acc[j][e] = 0.0f;

    const int arow = t >> 3;         // 0..31
    const int akq  = (t & 7) * 4;    // 0..28

    // ---- stage one 32 x (32k) A tile and 32k x 256 B tile, bf16, into LDS.
    // All global loads are issued before any LDS store so a single
    // s_wait_loadcnt covers the whole batch (keeps 9 loads in flight).
    auto stage = [&](int buf, int k0) {
        // A: one float4 per thread
        float4 av = *(const float4*)&A[(size_t)(m0 + arow) * K + k0 + akq];
        // B: 4 work items, each a PAIR of k-rows at the same 4 columns so the
        // two bf16 destined for Bs[col][2k],Bs[col][2k+1] pack into one b32.
        float4 b0[4], b1[4];
        #pragma unroll
        for (int w = 0; w < 4; ++w) {
            int id  = w * 256 + t;
            int kr2 = id >> 6;              // 0..15 (pair index)
            int c4  = id & 63;              // 0..63 (float4 column group)
            const float* base = &W[(size_t)(k0 + 2 * kr2) * N + n0 + c4 * 4];
            b0[w] = *(const float4*)base;
            b1[w] = *(const float4*)(base + N);
        }
        // stream weights ahead of the pipeline (global_prefetch_b8)
        if (k0 + 96 <= K)
            __builtin_prefetch(&W[(size_t)(k0 + 64 + (t >> 3)) * N + n0 + (t & 7) * 32], 0, 0);

        bf16x4 a4 = { (__bf16)av.x, (__bf16)av.y, (__bf16)av.z, (__bf16)av.w };
        *(bf16x4*)&As[buf][arow][akq] = a4;          // one ds_store_b64

        #pragma unroll
        for (int w = 0; w < 4; ++w) {
            int id  = w * 256 + t;
            int kr2 = id >> 6;
            int c4  = id & 63;
            const float* p0 = (const float*)&b0[w];
            const float* p1 = (const float*)&b1[w];
            #pragma unroll
            for (int u = 0; u < 4; ++u) {
                bf16x2 pv = { (__bf16)p0[u], (__bf16)p1[u] };
                *(bf16x2*)&Bs[buf][c4 * 4 + u][2 * kr2] = pv;   // ds_store_b32
            }
        }
    };

    // ---- one 32-k compute step on LDS buffer `cur` (compile-time constant)
    auto compute = [&](int cur) {
        // A fragment (16x32 bf16): row = lane%16,
        // elems 0..7 -> K = 8*hl + e ; elems 8..15 -> K = 8*hl + 16 + e
        bf16x8 alo = *(const bf16x8*)&As[cur][wm * 16 + r16][hl * 8];
        bf16x8 ahi = *(const bf16x8*)&As[cur][wm * 16 + r16][hl * 8 + 16];
        bf16x16 afrag;
        #pragma unroll
        for (int e = 0; e < 8; ++e) { afrag[e] = alo[e]; afrag[e + 8] = ahi[e]; }

        #pragma unroll
        for (int j = 0; j < 4; ++j) {
            // B fragment (32x16 bf16): col = lane%16, elems 0..7 -> K = 8*hl + e ; elems 8..15 -> K = 8*hl + 16 + e
            const __bf16* bp = &Bs[cur][wn * 64 + j * 16 + r16][hl * 8];
            bf16x8 b0 = *(const bf16x8*)bp;
            bf16x8 b1 = *(const bf16x8*)(bp + 16);
            bf16x16 bfrag;
            #pragma unroll
            for (int e = 0; e < 8; ++e) { bfrag[e] = b0[e]; bfrag[e + 8] = b1[e]; }

            acc[j] = __builtin_amdgcn_wmma_f32_16x16x32_bf16(
                false, afrag, false, bfrag, (short)0, acc[j], false, false);
        }
    };

    stage(0, 0);
    __syncthreads();

    const int ksteps = K >> 5;               // even for every call site
    for (int kt = 0; kt < ksteps; kt += 2) {
        stage(1, (kt + 1) << 5);             // fill buf1 while computing buf0
        compute(0);
        __syncthreads();
        if (kt + 2 < ksteps) stage(0, (kt + 2) << 5);
        compute(1);
        __syncthreads();
    }

    // ---- epilogue:  C/D layout: vgpr i -> M = (hl? 8+i : i), N = lane%16
    float sq = 0.0f;
    #pragma unroll
    for (int j = 0; j < 4; ++j) {
        const int col = n0 + wn * 64 + j * 16 + r16;
        const float bv = bias[col];
        #pragma unroll
        for (int i = 0; i < 8; ++i) {
            const int row = m0 + wm * 16 + (hl ? 8 + i : i);
            float v = acc[j][i] + bv;
            if (EPI == EPI_GELU)            v = gelu_tanh(v);
            else if (EPI == EPI_SOFTPLUS5)  v = softplus_f(v - 5.0f);
            else if (EPI == EPI_RESID_SQ) {
                sq += v * v;                       // recon = sum((out-x)^2)
                v  += resid[(size_t)row * N + col];
            }
            Out[(size_t)row * N + col] = v;
        }
    }
    if (EPI == EPI_RESID_SQ) {
        red[t] = sq;
        __syncthreads();
        for (int s = 128; s > 0; s >>= 1) {
            if (t < s) red[t] += red[t + s];
            __syncthreads();
        }
        if (t == 0) partial[blockIdx.y * gridDim.x + blockIdx.x] = red[0];
    }
}

// ---------------------------------------------------------------------------
// Haar DWT + transpose to (R,P)
// ---------------------------------------------------------------------------
__global__ __launch_bounds__(256) void
dwt_kernel(const float* __restrict__ x, float* __restrict__ cA,
           float* __restrict__ cD, float* __restrict__ seas,
           float* __restrict__ trend)
{
    int idx = blockIdx.x * 256 + threadIdx.x;
    if (idx >= NPC) return;
    int c = idx % C_;
    int p = (idx / C_) % P_;
    int b = idx / (C_ * P_);
    const float inv = 0.70710678118654752440f;
    float x0 = x[((size_t)b * L_ + 2 * p) * C_ + c];
    float x1 = x[((size_t)b * L_ + 2 * p + 1) * C_ + c];
    float a = (x0 + x1) * inv;
    float d = (x0 - x1) * inv;
    cA[idx] = a;
    cD[idx] = d;
    trend[((size_t)(b * C_ + c)) * P_ + p] = a;
    seas [((size_t)(b * C_ + c)) * P_ + p] = d;
}

// ---------------------------------------------------------------------------
// z = mu + std*eps, with fused per-block KL partial:
// kl = -0.5 * sum(1 + 2*log(std) - mu^2 - std^2)   (we store the inner sum)
// ---------------------------------------------------------------------------
__global__ __launch_bounds__(256) void
zkl_kernel(const float* __restrict__ mu, const float* __restrict__ stdv,
           const float* __restrict__ eps, float* __restrict__ z,
           float* __restrict__ klpart)
{
    __shared__ float red[256];
    int idx = blockIdx.x * 256 + threadIdx.x;
    float m = mu[idx];
    float s = stdv[idx];
    z[idx] = m + s * eps[idx];
    float kle = 1.0f + 2.0f * logf(s) - m * m - s * s;
    red[threadIdx.x] = kle;
    __syncthreads();
    for (int st = 128; st > 0; st >>= 1) {
        if (threadIdx.x < st) red[threadIdx.x] += red[threadIdx.x + st];
        __syncthreads();
    }
    if (threadIdx.x == 0) klpart[blockIdx.x] = red[0];
}

// ---------------------------------------------------------------------------
// Haar IDWT of (cA + trend_o, cD + seas_o) -> final_ (B,L,C) and y0 (R,L)
// ---------------------------------------------------------------------------
__global__ __launch_bounds__(256) void
idwt_kernel(const float* __restrict__ cA, const float* __restrict__ cD,
            const float* __restrict__ trend_o, const float* __restrict__ seas_o,
            float* __restrict__ final_, float* __restrict__ y0)
{
    int idx = blockIdx.x * 256 + threadIdx.x;
    if (idx >= NPC) return;
    int c = idx % C_;
    int p = (idx / C_) % P_;
    int b = idx / (C_ * P_);
    const float inv = 0.70710678118654752440f;
    size_t tp = ((size_t)(b * C_ + c)) * P_ + p;
    float a = cA[idx] + trend_o[tp];
    float d = cD[idx] + seas_o[tp];
    float f0 = (a + d) * inv;
    float f1 = (a - d) * inv;
    final_[((size_t)b * L_ + 2 * p) * C_ + c]     = f0;
    final_[((size_t)b * L_ + 2 * p + 1) * C_ + c] = f1;
    y0[((size_t)(b * C_ + c)) * L_ + 2 * p]     = f0;
    y0[((size_t)(b * C_ + c)) * L_ + 2 * p + 1] = f1;
}

// ---------------------------------------------------------------------------
// dec = transpose(y4 (R,P)) -> (B,P,C); slice [:, -P:, :] is the whole tensor
// ---------------------------------------------------------------------------
__global__ __launch_bounds__(256) void
dec_kernel(const float* __restrict__ y4, float* __restrict__ out0)
{
    int idx = blockIdx.x * 256 + threadIdx.x;
    if (idx >= NPC) return;
    int c = idx % C_;
    int p = (idx / C_) % P_;
    int b = idx / (C_ * P_);
    out0[idx] = y4[((size_t)(b * C_ + c)) * P_ + p];
}

// ---------------------------------------------------------------------------
// Deterministic fixed-order loss combiner (single block).
// part layout: recon_s @0 (21) | kl_s @32 (672) | recon_t @704 (21) | kl_t @736 (672)
// loss = 1e-4 * (recon_s + recon_t + 0.001 * (-0.5) * (klsum_s + klsum_t))
// ---------------------------------------------------------------------------
__global__ __launch_bounds__(256) void
loss_kernel(const float* __restrict__ part, float* __restrict__ out_loss)
{
    __shared__ float red[256];
    float acc = 0.0f;
    for (int i = threadIdx.x; i < 21; i += 256)
        acc += part[i] + part[704 + i];
    float kl = 0.0f;
    for (int i = threadIdx.x; i < 672; i += 256)
        kl += part[32 + i] + part[736 + i];
    acc += 0.001f * (-0.5f) * kl;
    red[threadIdx.x] = acc;
    __syncthreads();
    for (int s = 128; s > 0; s >>= 1) {
        if (threadIdx.x < s) red[threadIdx.x] += red[threadIdx.x + s];
        __syncthreads();
    }
    if (threadIdx.x == 0) out_loss[0] = red[0] * 1.0e-4f;
}

// ---------------------------------------------------------------------------
// Host side
// ---------------------------------------------------------------------------
static void launch_gemm(int epi, const float* A, const float* W, const float* bias,
                        float* Out, int K, int N, const float* resid, float* part,
                        hipStream_t s)
{
    dim3 g(N / 256, R_ / 32), b(256);
    switch (epi) {
    case EPI_NONE:
        gemm_bf16_wmma<EPI_NONE><<<g, b, 0, s>>>(A, W, bias, Out, K, N, resid, part);
        break;
    case EPI_GELU:
        gemm_bf16_wmma<EPI_GELU><<<g, b, 0, s>>>(A, W, bias, Out, K, N, resid, part);
        break;
    case EPI_SOFTPLUS5:
        gemm_bf16_wmma<EPI_SOFTPLUS5><<<g, b, 0, s>>>(A, W, bias, Out, K, N, resid, part);
        break;
    case EPI_RESID_SQ:
        gemm_bf16_wmma<EPI_RESID_SQ><<<g, b, 0, s>>>(A, W, bias, Out, K, N, resid, part);
        break;
    }
}

extern "C" void kernel_launch(void* const* d_in, const int* in_sizes, int n_in,
                              void* d_out, int out_size, void* d_ws, size_t ws_size,
                              hipStream_t stream)
{
    const float* x_enc = (const float*)d_in[0];
    const float* eps_s = (const float*)d_in[1];
    const float* eps_t = (const float*)d_in[2];
    const float* sw1  = (const float*)d_in[3];  const float* sb1  = (const float*)d_in[4];
    const float* sw2  = (const float*)d_in[5];  const float* sb2  = (const float*)d_in[6];
    const float* smuw = (const float*)d_in[7];  const float* smub = (const float*)d_in[8];
    const float* sstdw= (const float*)d_in[9];  const float* sstdb= (const float*)d_in[10];
    const float* sdecw= (const float*)d_in[11]; const float* sdecb= (const float*)d_in[12];
    const float* tw1  = (const float*)d_in[13]; const float* tb1  = (const float*)d_in[14];
    const float* tw2  = (const float*)d_in[15]; const float* tb2  = (const float*)d_in[16];
    const float* tmuw = (const float*)d_in[17]; const float* tmub = (const float*)d_in[18];
    const float* tstdw= (const float*)d_in[19]; const float* tstdb= (const float*)d_in[20];
    const float* tdecw= (const float*)d_in[21]; const float* tdecb= (const float*)d_in[22];
    const float* ow1 = (const float*)d_in[23];  const float* ob1 = (const float*)d_in[24];
    const float* ow2 = (const float*)d_in[25];  const float* ob2 = (const float*)d_in[26];
    const float* ow3 = (const float*)d_in[27];  const float* ob3 = (const float*)d_in[28];
    const float* ow4 = (const float*)d_in[29];  const float* ob4 = (const float*)d_in[30];

    float* out0   = (float*)d_out;            // dec slice  (B,P,C)  = 172032
    float* outL   = out0 + NPC;               // scalar loss
    float* final_ = out0 + NPC + 1;           // final_     (B,L,C)  = 344064

    // workspace layout (floats)
    float* ws = (float*)d_ws;
    float* cA     = ws;                 // 172032
    float* cD     = cA     + NPC;       // 172032
    float* seas   = cD     + NPC;       // 172032
    float* trend  = seas   + NPC;       // 172032
    float* t_h1   = trend  + NPC;       // 172032
    float* t_h    = t_h1   + NPC;       // 172032
    float* t_mu   = t_h    + NPC;       // 172032
    float* t_std  = t_mu   + NPC;       // 172032
    float* t_z    = t_std  + NPC;       // 172032
    float* seas_o = t_z    + NPC;       // 172032
    float* trend_o= seas_o + NPC;       // 172032
    float* y0     = trend_o+ NPC;       // R*512  = 344064
    float* a1     = y0     + (size_t)R_ * L_;        // R*5120
    float* a2     = a1     + (size_t)R_ * 2 * 2560;  // R*10240
    float* a3     = a2     + (size_t)R_ * 4 * 2560;  // R*20480
    float* y4     = a3     + (size_t)R_ * 8 * 2560;  // R*256
    float* part   = y4     + (size_t)R_ * P_;        // 1408 partials
    float* recon_s = part;        // 21 used
    float* kl_s    = part + 32;   // 672
    float* recon_t = part + 704;  // 21 used
    float* kl_t    = part + 736;  // 672

    const int eb = NPC / 256;     // 672 blocks for elementwise kernels

    // 1) DWT + transposes
    dwt_kernel<<<eb, 256, 0, stream>>>(x_enc, cA, cD, seas, trend);

    // 2) seasonal IB block
    launch_gemm(EPI_GELU,      seas, sw1,  sb1,  t_h1, P_, P_, nullptr, nullptr, stream);
    launch_gemm(EPI_NONE,      t_h1, sw2,  sb2,  t_h,  P_, P_, nullptr, nullptr, stream);
    launch_gemm(EPI_NONE,      t_h,  smuw, smub, t_mu, P_, P_, nullptr, nullptr, stream);
    launch_gemm(EPI_SOFTPLUS5, t_h,  sstdw,sstdb,t_std,P_, P_, nullptr, nullptr, stream);
    zkl_kernel<<<eb, 256, 0, stream>>>(t_mu, t_std, eps_s, t_z, kl_s);
    launch_gemm(EPI_RESID_SQ,  t_z,  sdecw,sdecb,seas_o,P_, P_, seas,  recon_s, stream);

    // 3) trend IB block
    launch_gemm(EPI_GELU,      trend, tw1,  tb1,  t_h1, P_, P_, nullptr, nullptr, stream);
    launch_gemm(EPI_NONE,      t_h1,  tw2,  tb2,  t_h,  P_, P_, nullptr, nullptr, stream);
    launch_gemm(EPI_NONE,      t_h,   tmuw, tmub, t_mu, P_, P_, nullptr, nullptr, stream);
    launch_gemm(EPI_SOFTPLUS5, t_h,   tstdw,tstdb,t_std,P_, P_, nullptr, nullptr, stream);
    zkl_kernel<<<eb, 256, 0, stream>>>(t_mu, t_std, eps_t, t_z, kl_t);
    launch_gemm(EPI_RESID_SQ,  t_z,   tdecw,tdecb,trend_o,P_, P_, trend, recon_t, stream);

    // 4) IDWT -> final_ (output #3) and y0 (R x 512)
    idwt_kernel<<<eb, 256, 0, stream>>>(cA, cD, trend_o, seas_o, final_, y0);

    // 5) the weight-streaming MLP (HBM bound; ~1.07 GB of weights)
    launch_gemm(EPI_NONE, y0, ow1, ob1, a1, 512,   5120,  nullptr, nullptr, stream);
    launch_gemm(EPI_NONE, a1, ow2, ob2, a2, 5120,  10240, nullptr, nullptr, stream);
    launch_gemm(EPI_NONE, a2, ow3, ob3, a3, 10240, 20480, nullptr, nullptr, stream);
    launch_gemm(EPI_NONE, a3, ow4, ob4, y4, 20480, 256,   nullptr, nullptr, stream);

    // 6) dec transpose (output #1) and loss (output #2)
    dec_kernel<<<eb, 256, 0, stream>>>(y4, out0);
    loss_kernel<<<1, 256, 0, stream>>>(part, outL);

    (void)in_sizes; (void)n_in; (void)out_size; (void)ws_size;
}
